// Kattention_51393578664655
// MI455X (gfx1250) — matrix operand order repaired and checked by reading, not compiled, not run on hardware
//
#include <hip/hip_runtime.h>
#include <hip/hip_bf16.h>

// ---------------------------------------------------------------------------
// Fused hard-Gumbel-softmax attention block for MI455X (gfx1250, wave32).
//   K1: qkv = x @ w_attn + b   (bf16 WMMA, f32 accum) -> q,k,v [B,H,T,64] bf16
//   K2: fused scores + causal mask + hard gumbel softmax (online max/argmax/
//       sum-exp). K tiles stream through the Tensor Data Mover into LDS with
//       hardware padding, double-buffered on TENSORcnt.
//   K3: out = y @ w_proj + b_proj. A tile staged by TDM (bf16), B converted.
// ---------------------------------------------------------------------------

typedef __attribute__((ext_vector_type(16))) __bf16        v16bf;
typedef __attribute__((ext_vector_type(8)))  float         v8f;
typedef __attribute__((ext_vector_type(8)))  unsigned int  v8u;
typedef __attribute__((ext_vector_type(4)))  unsigned int  v4u;
typedef __attribute__((ext_vector_type(8)))  int           v8i_;
typedef __attribute__((ext_vector_type(4)))  int           v4i_;

union FragBF { v8u u; v16bf b; };

__device__ __forceinline__ unsigned short f2bf(float f) {
  unsigned int u = __float_as_uint(f);
  unsigned int r = u + 0x7FFFu + ((u >> 16) & 1u);
  return (unsigned short)(r >> 16);
}
__device__ __forceinline__ unsigned int pack2bf(float lo, float hi) {
  return (unsigned int)f2bf(lo) | ((unsigned int)f2bf(hi) << 16);
}

// --- Tensor Data Mover: 2-D tile load (D# groups per CDNA5 ISA) --
// cfg packs group1 dword0: data_size[17:16], pad_enable[20], pad_interval[24:22],
// pad_amount[31:25]. Dims/strides in data_size units. 6-arg builtin form
// (this toolchain): (g0 u32x4, g1 i32x8, g2 i32x4, g3 i32x4, i32x8, cpol).
__device__ __forceinline__ void tdm_load_2d(unsigned ldsOff, unsigned long ga,
                                            unsigned cfg, unsigned td0,
                                            unsigned td1, unsigned tile0,
                                            unsigned tile1, unsigned long s0) {
  v4u g0;
  g0[0] = 1u;                                   // count=1, user descriptor
  g0[1] = (unsigned)__builtin_amdgcn_readfirstlane((int)ldsOff);
  g0[2] = (unsigned)__builtin_amdgcn_readfirstlane((int)(unsigned)ga);
  g0[3] = ((unsigned)__builtin_amdgcn_readfirstlane((int)(unsigned)(ga >> 32)) & 0x1ffffffu)
          | (2u << 30);                          // type=2 ("image")
  v8i_ g1;
  g1[0] = (int)cfg;
  g1[1] = (int)((td0 & 0xffffu) << 16);                                   // tensor_dim0 lo
  g1[2] = (int)(((td0 >> 16) & 0xffffu) | ((td1 & 0xffffu) << 16));       // dim0 hi | dim1 lo
  g1[3] = (int)(((td1 >> 16) & 0xffffu) | ((tile0 & 0xffffu) << 16));     // dim1 hi | tile0
  g1[4] = (int)(tile1 & 0xffffu);                                         // tile1 (tile2=0)
  g1[5] = (int)(unsigned)(s0 & 0xffffffffu);                              // dim0_stride lo
  g1[6] = (int)(unsigned)((s0 >> 32) & 0xffffu);                          // stride hi
  g1[7] = 0;
  v4i_ z4 = {0, 0, 0, 0};
  v8i_ z8 = {0, 0, 0, 0, 0, 0, 0, 0};
  __builtin_amdgcn_tensor_load_to_lds(g0, g1, z4, z4, z8, 0);
}

#define PA 21  // A-tile LDS pitch in dwords (16 data dwords + pad; conflict-free)
#define PB 19  // B-tile LDS pitch in dwords (16 data dwords + pad; conflict-free)

// ---------------------------------------------------------------------------
// K1: qkv projection. Grid: (2304/128) * (4096/128) = 18*32 blocks, 256 thr.
// ---------------------------------------------------------------------------
__global__ __launch_bounds__(256)
void qkv_gemm_kernel(const float* __restrict__ x,      // [4096,768]
                     const float* __restrict__ w,      // [768,2304]
                     const float* __restrict__ bias,   // [2304]
                     unsigned short* __restrict__ qh,
                     unsigned short* __restrict__ kh,
                     unsigned short* __restrict__ vh) {
  __shared__ unsigned int   AsU[128 * PA];
  __shared__ unsigned short BsH[128 * 2 * PB];
  unsigned int* BsU32 = (unsigned int*)BsH;

  const int tid  = threadIdx.x;
  const int lane = tid & 31, wid = tid >> 5;
  const int nt = blockIdx.x % 18, mt = blockIdx.x / 18;
  const int m0 = mt * 128, n0 = nt * 128;
  const int wave_m = (wid >> 2) * 64, wave_n = (wid & 3) * 32;
  const int khalf = lane >> 4, l15 = lane & 15;

  const v8f vzero = {0.f,0.f,0.f,0.f,0.f,0.f,0.f,0.f};
  v8f acc[4][2];
#pragma unroll
  for (int i = 0; i < 4; i++) { acc[i][0] = vzero; acc[i][1] = vzero; }

  for (int k0 = 0; k0 < 768; k0 += 32) {
    __syncthreads();
    for (int i = tid; i < 128 * 16; i += 256) {  // A fp32 -> bf16 pairs
      int m = i >> 4, kd = i & 15;
      const float* p = &x[(m0 + m) * 768 + k0 + kd * 2];
      AsU[m * PA + kd] = pack2bf(p[0], p[1]);
    }
    for (int i = tid; i < 32 * 128; i += 256) {  // B transposed fp32 -> bf16
      int k = i >> 7, n = i & 127;
      BsH[n * (2 * PB) + k] = f2bf(w[(k0 + k) * 2304 + n0 + n]);
    }
    __syncthreads();

    FragBF b0, b1;
#pragma unroll
    for (int v = 0; v < 8; v++) {
      int d = (v & 3) + 4 * khalf + 8 * (v >> 2);
      b0.u[v] = BsU32[(wave_n      + l15) * PB + d];
      b1.u[v] = BsU32[(wave_n + 16 + l15) * PB + d];
    }
#pragma unroll
    for (int i = 0; i < 4; i++) {
      FragBF a;
#pragma unroll
      for (int v = 0; v < 8; v++) {
        int d = (v & 3) + 4 * khalf + 8 * (v >> 2);
        a.u[v] = AsU[(wave_m + i * 16 + l15) * PA + d];
      }
      acc[i][0] = __builtin_amdgcn_wmma_f32_16x16x32_bf16(false, a.b, false, b0.b, (short)0, acc[i][0], false, false);
      acc[i][1] = __builtin_amdgcn_wmma_f32_16x16x32_bf16(false, a.b, false, b1.b, (short)0, acc[i][1], false, false);
    }
  }

#pragma unroll
  for (int i = 0; i < 4; i++)
#pragma unroll
    for (int j = 0; j < 2; j++)
#pragma unroll
      for (int r = 0; r < 8; r++) {
        int gm = m0 + wave_m + i * 16 + r + 8 * khalf;
        int gn = n0 + wave_n + j * 16 + l15;
        float val = acc[i][j][r] + bias[gn];
        int which = gn / 768;
        int cc = gn - which * 768;
        int h = cc >> 6, d = cc & 63;
        int bb = gm >> 11, t = gm & 2047;
        unsigned short* dst = (which == 0) ? qh : ((which == 1) ? kh : vh);
        dst[((((bb * 12 + h) << 11) + t) << 6) + d] = f2bf(val);
      }
}

// ---------------------------------------------------------------------------
// K2: fused causal hard-gumbel attention. One wave owns a 16-row q tile.
// K tiles arrive via TDM (double-buffered on TENSORcnt); hardware pads rows
// to a 33-dword pitch for conflict-free fragment reads.
// Grid: 24*128/8 = 384 blocks x 256 threads. Longest tiles launch first.
// ---------------------------------------------------------------------------
#define KPITCH 33   // 32 data dwords + 1 pad dword per 16-qword row (TDM pad)

__global__ __launch_bounds__(256)
void gumbel_attn_kernel(const unsigned int*   __restrict__ qu,  // [BH,2048,32] dw
                        const unsigned short* __restrict__ kh,  // [BH,2048,64] bf16
                        const unsigned int*   __restrict__ vu,
                        const float*          __restrict__ gum, // [BH,2048,2048]
                        unsigned int*         __restrict__ yu)  // [4096,384] dw
{
  __shared__ unsigned int kbuf[8][2][16 * KPITCH];   // per-wave double buffer

  const int tid = threadIdx.x, lane = tid & 31, wid = tid >> 5;
  const int qt    = 3071 - (blockIdx.x * 8 + wid);  // reverse for balance
  const int bh    = qt >> 7;
  const int qtile = qt & 127;
  const int q0    = qtile << 4;
  const int khalf = lane >> 4, l15 = lane & 15;
  const int bb = bh / 12, h = bh - bb * 12;

  const unsigned myLds[2] = {
      (unsigned)(size_t)(void*)&kbuf[wid][0][0],
      (unsigned)(size_t)(void*)&kbuf[wid][1][0] };
  const unsigned long kByteBase = (unsigned long)(size_t)kh;

  // TDM 2-D descriptor: 2B elems, 64x16 tile, row stride 64 elems,
  // pad_interval=4 (16 qwords) + pad_amount=1 dword  ->  LDS pitch 33 dwords.
  auto issueK = [&](int ktile, unsigned ldsOff) {
    unsigned long ga = kByteBase +
        (((unsigned long)bh * 2048ul + (unsigned long)(ktile << 4)) << 7); // *64*2B
    tdm_load_2d(ldsOff, ga, 0x01110000u, 64u, 0x100000u, 64u, 16u, 64ul);
  };

  // Q fragments (A matrix, K = head-dim 0..31 and 32..63)
  FragBF aq0, aq1;
  {
    long base = ((long)bh * 2048 + q0 + l15) * 32;
#pragma unroll
    for (int v = 0; v < 8; v++) {
      int d = (v & 3) + 4 * khalf + 8 * (v >> 2);
      aq0.u[v] = qu[base + d];
      aq1.u[v] = qu[base + 16 + d];
    }
  }

  float rm[8], rd[8];
  int   ka[8];
#pragma unroll
  for (int r = 0; r < 8; r++) { rm[r] = -3.4e38f; rd[r] = 0.0f; ka[r] = 0; }

  const float scale  = 0.125f;  // 1/sqrt(64)
  const float invtau = 2.0f;    // 1/TAU
  const long  grow   = (long)bh * 2048L * 2048L;
  const v8f   vzero  = {0.f,0.f,0.f,0.f,0.f,0.f,0.f,0.f};

  issueK(0, myLds[0]);                       // prologue DMA

  for (int kt = 0; kt <= qtile; kt++) {
    const int k0 = kt << 4;
    if (kt < qtile) {
      issueK(kt + 1, myLds[(kt + 1) & 1]);   // prefetch next tile via TDM
      __builtin_amdgcn_s_wait_tensorcnt(1);  // current tile landed
    } else {
      __builtin_amdgcn_s_wait_tensorcnt(0);
    }
    const unsigned int* kb = kbuf[wid][kt & 1];

    FragBF bk0, bk1;
#pragma unroll
    for (int v = 0; v < 8; v++) {
      int d = (v & 3) + 4 * khalf + 8 * (v >> 2);
      bk0.u[v] = kb[l15 * KPITCH + d];
      bk1.u[v] = kb[l15 * KPITCH + 16 + d];
    }
    v8f s = vzero;
    s = __builtin_amdgcn_wmma_f32_16x16x32_bf16(false, aq0.b, false, bk0.b, (short)0, s, false, false);
    s = __builtin_amdgcn_wmma_f32_16x16x32_bf16(false, aq1.b, false, bk1.b, (short)0, s, false, false);

#pragma unroll
    for (int r = 0; r < 8; r++) {
      const int qrow = q0 + r + 8 * khalf;
      const int kcol = k0 + l15;
      const float* gp = &gum[grow + (long)qrow * 2048 + kcol];
      if (kt < qtile) __builtin_prefetch(gp + 16, 0, 1);  // next gumbel tile
      float g = *gp;
      float z = (kcol <= qrow) ? (s[r] * scale + g) * invtau : -3.0e9f;

      float bm = z; int bi = kcol;
#pragma unroll
      for (int off = 1; off < 16; off <<= 1) {
        float om = __shfl_xor(bm, off, 32);
        int   oi = __shfl_xor(bi, off, 32);
        if (om > bm || (om == bm && oi < bi)) { bm = om; bi = oi; }
      }
      float nm = fmaxf(rm[r], bm);
      float e  = __expf(z - nm);
#pragma unroll
      for (int off = 1; off < 16; off <<= 1) e += __shfl_xor(e, off, 32);
      rd[r] = rd[r] * __expf(rm[r] - nm) + e;
      if (bm > rm[r]) ka[r] = bi;   // strict >: earliest index wins ties
      rm[r] = nm;
    }
  }

  // Straight-through forward: row -> c * v[k*], c = (1-p)+p, p = 1/denom
#pragma unroll
  for (int r = 0; r < 8; r++) {
    const int qrow = q0 + r + 8 * khalf;
    float p = 1.0f / rd[r];
    float c = (1.0f - p) + p;
    long vbase = ((long)bh * 2048 + ka[r]) * 32;
    unsigned int v0 = vu[vbase + l15];
    unsigned int v1 = vu[vbase + 16 + l15];
    unsigned int o0 = pack2bf(__uint_as_float(v0 << 16) * c,
                              __uint_as_float(v0 & 0xffff0000u) * c);
    unsigned int o1 = pack2bf(__uint_as_float(v1 << 16) * c,
                              __uint_as_float(v1 & 0xffff0000u) * c);
    long ybase = (long)(bb * 2048 + qrow) * 384 + h * 32 + l15;
    yu[ybase]      = o0;
    yu[ybase + 16] = o1;
  }
}

// ---------------------------------------------------------------------------
// K3: output projection. A tile (bf16) staged by TDM with hardware padding to
// the 21-dword pitch; B tile converted fp32->bf16 by all threads in parallel
// with the DMA. Grid: (768/128)*(4096/128) = 6*32 blocks, 256 thr.
// ---------------------------------------------------------------------------
__global__ __launch_bounds__(256)
void proj_gemm_kernel(const unsigned int* __restrict__ yu,   // [4096,384] dw bf16
                      const float* __restrict__ w,           // [768,768]
                      const float* __restrict__ bias,        // [768]
                      float* __restrict__ out) {             // [4096,768]
  __shared__ unsigned int   AsU[128 * PA];
  __shared__ unsigned short BsH[128 * 2 * PB];
  unsigned int* BsU32 = (unsigned int*)BsH;

  const int tid  = threadIdx.x;
  const int lane = tid & 31, wid = tid >> 5;
  const int nt = blockIdx.x % 6, mt = blockIdx.x / 6;
  const int m0 = mt * 128, n0 = nt * 128;
  const int wave_m = (wid >> 2) * 64, wave_n = (wid & 3) * 32;
  const int khalf = lane >> 4, l15 = lane & 15;

  const unsigned asOff = (unsigned)(size_t)(void*)&AsU[0];
  const unsigned long yByteBase = (unsigned long)(size_t)yu;

  const v8f vzero = {0.f,0.f,0.f,0.f,0.f,0.f,0.f,0.f};
  v8f acc[4][2];
#pragma unroll
  for (int i = 0; i < 4; i++) { acc[i][0] = vzero; acc[i][1] = vzero; }

  for (int k0 = 0; k0 < 768; k0 += 32) {
    __syncthreads();
    if (wid == 0) {
      // A tile: 32x128 bf16, tensor row = 768 elems; pad_interval=3 (8 qwords)
      // + pad_amount=5 dwords -> LDS pitch 21 dwords (matches PA).
      unsigned long ga = yByteBase +
          (((unsigned long)m0 * 384ul + (unsigned long)(k0 >> 1)) << 2);
      tdm_load_2d(asOff, ga, 0x08D10000u, 768u, 0x100000u, 32u, 128u, 768ul);
    }
    for (int i = tid; i < 32 * 128; i += 256) {   // B fp32 -> bf16, transposed
      int k = i >> 7, n = i & 127;
      BsH[n * (2 * PB) + k] = f2bf(w[(k0 + k) * 768 + n0 + n]);
    }
    if (wid == 0) __builtin_amdgcn_s_wait_tensorcnt(0);
    __syncthreads();

    FragBF b0, b1;
#pragma unroll
    for (int v = 0; v < 8; v++) {
      int d = (v & 3) + 4 * khalf + 8 * (v >> 2);
      b0.u[v] = BsU32[(wave_n      + l15) * PB + d];
      b1.u[v] = BsU32[(wave_n + 16 + l15) * PB + d];
    }
#pragma unroll
    for (int i = 0; i < 4; i++) {
      FragBF a;
#pragma unroll
      for (int v = 0; v < 8; v++) {
        int d = (v & 3) + 4 * khalf + 8 * (v >> 2);
        a.u[v] = AsU[(wave_m + i * 16 + l15) * PA + d];
      }
      acc[i][0] = __builtin_amdgcn_wmma_f32_16x16x32_bf16(false, a.b, false, b0.b, (short)0, acc[i][0], false, false);
      acc[i][1] = __builtin_amdgcn_wmma_f32_16x16x32_bf16(false, a.b, false, b1.b, (short)0, acc[i][1], false, false);
    }
  }

#pragma unroll
  for (int i = 0; i < 4; i++)
#pragma unroll
    for (int j = 0; j < 2; j++)
#pragma unroll
      for (int r = 0; r < 8; r++) {
        int gm = m0 + wave_m + i * 16 + r + 8 * khalf;
        int gn = n0 + wave_n + j * 16 + l15;
        out[gm * 768 + gn] = acc[i][j][r] + bias[gn];
      }
}

// ---------------------------------------------------------------------------
extern "C" void kernel_launch(void* const* d_in, const int* in_sizes, int n_in,
                              void* d_out, int out_size, void* d_ws, size_t ws_size,
                              hipStream_t stream) {
  const float* x      = (const float*)d_in[0];   // [2,2048,768]
  const float* w_attn = (const float*)d_in[1];   // [768,2304]
  const float* b_attn = (const float*)d_in[2];   // [2304]
  const float* w_proj = (const float*)d_in[3];   // [768,768]
  const float* b_proj = (const float*)d_in[4];   // [768]
  const float* gumbel = (const float*)d_in[5];   // [2,12,2048,2048]

  // workspace: q,k,v [24,2048,64] bf16 + y [4096,768] bf16 = 24 MB
  const size_t QKV = (size_t)24 * 2048 * 64;
  unsigned short* qh = (unsigned short*)d_ws;
  unsigned short* kh = qh + QKV;
  unsigned short* vh = kh + QKV;
  unsigned short* yh = vh + QKV;

  qkv_gemm_kernel<<<dim3(18 * 32), dim3(256), 0, stream>>>(
      x, w_attn, b_attn, qh, kh, vh);

  gumbel_attn_kernel<<<dim3(384), dim3(256), 0, stream>>>(
      (const unsigned int*)qh, kh,
      (const unsigned int*)vh, gumbel, (unsigned int*)yh);

  proj_gemm_kernel<<<dim3(6 * 32), dim3(256), 0, stream>>>(
      (const unsigned int*)yh, w_proj, b_proj, (float*)d_out);
}
